// LatentAttention_45114336477260
// MI455X (gfx1250) — matrix operand order repaired and checked by reading, not compiled, not run on hardware
//
#include <hip/hip_runtime.h>
#include <hip/hip_bf16.h>

// ---------------------------------------------------------------------------
// MLA latent attention for MI455X (gfx1250, wave32, WMMA).
// fp32 inputs are converted to bf16; all GEMMs + attention matmuls use
// v_wmma_f32_16x16x32_bf16 with fp32 accumulation.
// Data movement uses the CDNA5-specific paths:
//  - GEMM A tiles:      global_load_async_to_lds_b128   (ASYNCcnt)
//  - attention K tiles: tensor_load_to_lds (TDM, D# descriptor, TENSORcnt)
// ---------------------------------------------------------------------------

typedef __bf16 bf16;
typedef __attribute__((ext_vector_type(8)))  __bf16 v8bf;
typedef __attribute__((ext_vector_type(16))) __bf16 v16bf;
typedef __attribute__((ext_vector_type(8)))  float  v8f;
typedef __attribute__((ext_vector_type(4)))  unsigned int v4u;
typedef __attribute__((ext_vector_type(4)))  int v4i;
typedef __attribute__((ext_vector_type(8)))  int v8i;

#define DEV static __device__ __forceinline__

DEV bf16 to_bf16(float f) {
  unsigned u = __builtin_bit_cast(unsigned, f);
  u += 0x7FFFu + ((u >> 16) & 1u);               // round-to-nearest-even
  unsigned short h = (unsigned short)(u >> 16);
  return __builtin_bit_cast(bf16, h);
}
DEV float from_bf16(bf16 b) {
  unsigned short h = __builtin_bit_cast(unsigned short, b);
  unsigned u = ((unsigned)h) << 16;
  return __builtin_bit_cast(float, u);
}
DEV v16bf comb(v8bf lo, v8bf hi) {
  v16bf r;
#pragma unroll
  for (int i = 0; i < 8; ++i) { r[i] = lo[i]; r[i + 8] = hi[i]; }
  return r;
}
DEV v8f vzero8f() {
  v8f z;
#pragma unroll
  for (int i = 0; i < 8; ++i) z[i] = 0.0f;
  return z;
}
DEV v8bf vzero8bf() {
  v8bf z; bf16 b = __builtin_bit_cast(bf16, (unsigned short)0);
#pragma unroll
  for (int i = 0; i < 8; ++i) z[i] = b;
  return z;
}
DEV v8f wmma_bf16(v16bf a, v16bf b, v8f c) {
  // (neg_a, A, neg_b, B, c_mod, C, reuse_a, reuse_b)
  return __builtin_amdgcn_wmma_f32_16x16x32_bf16(false, a, false, b, (short)0, c,
                                                 false, false);
}

// ---- CDNA5 async global -> LDS copy (ASYNCcnt tracked) --------------------
DEV void async_g2l_b128(const void* lds_dst, const void* gsrc) {
  unsigned ldsoff = (unsigned)(uintptr_t)lds_dst;          // LDS addr = low 32b
  unsigned long long ga = (unsigned long long)(uintptr_t)gsrc;
  asm volatile("global_load_async_to_lds_b128 %0, %1, off"
               :: "v"(ldsoff), "v"(ga) : "memory");
}
DEV void wait_asynccnt0() { asm volatile("s_wait_asynccnt 0x0" ::: "memory"); }

// ---- CDNA5 Tensor Data Mover: 2D tile global -> LDS with row padding ------
// Builds a D# (groups 0/1 per ISA 8.3/8.4) for a 2D tile of 2-byte elements:
//   tile_d0 elems per row, tile_d1 rows, row stride stride_d0 elems,
//   LDS destination padded by `pad_amount` code after each `pad_interval`
//   code worth of stored data (codes per D# encoding).
DEV void tdm_load_2d(const void* lds_dst, const void* gsrc,
                     unsigned tile_d0, unsigned tile_d1, unsigned stride_d0,
                     unsigned pad_interval_code, unsigned pad_amount_code) {
  unsigned long long ga = (unsigned long long)(uintptr_t)gsrc;
  v4u g0;
  g0[0] = 1u;                                           // count=1, user mode
  g0[1] = (unsigned)(uintptr_t)lds_dst;                 // lds_addr (bytes)
  g0[2] = (unsigned)(ga & 0xFFFFFFFFu);                 // global_addr[31:0]
  g0[3] = (unsigned)((ga >> 32) & 0x01FFFFFFu) | (2u << 30);  // addr hi, type=2
  v8i g1;
  g1[0] = (int)((1u << 16)                              // data_size=1 (2B)
                | (1u << 20)                            // pad_enable
                | (pad_interval_code << 22)
                | (pad_amount_code << 25));
  g1[1] = (int)((tile_d0 & 0xFFFFu) << 16);             // tensor_dim0 lo16
  g1[2] = (int)(((tile_d0 >> 16) & 0xFFFFu)             // tensor_dim0 hi16
                | ((tile_d1 & 0xFFFFu) << 16));         // tensor_dim1 lo16
  g1[3] = (int)(((tile_d1 >> 16) & 0xFFFFu)             // tensor_dim1 hi16
                | ((tile_d0 & 0xFFFFu) << 16));         // tile_dim0
  g1[4] = (int)(tile_d1 & 0xFFFFu);                     // tile_dim1, tile_dim2=0
  g1[5] = (int)stride_d0;                               // dim0 stride lo32
  g1[6] = 0;                                            // stride hi, dim1 stride
  g1[7] = 0;
  v4i gz4; gz4[0] = 0; gz4[1] = 0; gz4[2] = 0; gz4[3] = 0;
#if defined(__clang_major__) && __clang_major__ >= 23
  v8i gz8;
#pragma unroll
  for (int i = 0; i < 8; ++i) gz8[i] = 0;
  __builtin_amdgcn_tensor_load_to_lds(g0, g1, gz4, gz4, gz8, 0);
#else
  __builtin_amdgcn_tensor_load_to_lds(g0, g1, gz4, gz4, 0);
#endif
}

// ---------------------------------------------------------------------------
// fp32 -> bf16 convert
// ---------------------------------------------------------------------------
__global__ void cvt_f32_bf16(const float* __restrict__ in, bf16* __restrict__ out,
                             size_t n) {
  size_t i = (size_t)blockIdx.x * blockDim.x + threadIdx.x;
  size_t stride = (size_t)gridDim.x * blockDim.x;
  for (; i < n; i += stride) out[i] = to_bf16(in[i]);
}

// ---------------------------------------------------------------------------
// bf16 GEMM: C[M,N] = A[M,K] * B[K,N]; A,B row-major bf16; C fp32 or bf16.
// Workgroup tile 128x128, 8 waves (4 in M x 2 in N), wave tile 32x64, K step 32.
// A tile staged with async global->LDS copies; B tile staged transposed.
// M multiple of 128, K multiple of 32; N guarded.
// ---------------------------------------------------------------------------
DEV void store_c(float* C, size_t idx, float v) { C[idx] = v; }
DEV void store_c(bf16* C, size_t idx, float v) { C[idx] = to_bf16(v); }

template <typename OutT>
__global__ __launch_bounds__(256) void gemm_bf16_kernel(
    const bf16* __restrict__ A, const bf16* __restrict__ B, OutT* __restrict__ C,
    int M, int N, int K) {
  __shared__ bf16 As[128 * 32];   // [row][k], stride 32
  __shared__ bf16 Bs[128 * 32];   // transposed: [n][k], stride 32
  const int tid = threadIdx.x;
  const int lane = tid & 31;
  const int wid = tid >> 5;
  const int ln = lane & 15;
  const int hl = lane >> 4;       // half of wave (0/1)
  const int wm = wid & 3;         // wave row  (0..3)
  const int wn = wid >> 2;        // wave col  (0..1)
  const int m0 = blockIdx.y * 128;
  const int n0 = blockIdx.x * 128;

  v8f acc[2][4];
#pragma unroll
  for (int mf = 0; mf < 2; ++mf)
#pragma unroll
    for (int nf = 0; nf < 4; ++nf) acc[mf][nf] = vzero8f();

  for (int k0 = 0; k0 < K; k0 += 32) {
    // async-stage A tile: 128x32 bf16 (512 chunks of 8), no VGPR round-trip
#pragma unroll
    for (int it = 0; it < 2; ++it) {
      int c = tid + it * 256;
      int r = c >> 2, kk = (c & 3) * 8;
      async_g2l_b128(&As[r * 32 + kk], A + (size_t)(m0 + r) * K + k0 + kk);
    }
    // prefetch next A tile into caches while this iteration computes
    if (k0 + 32 < K)
      __builtin_prefetch(A + (size_t)(m0 + (tid >> 1)) * K + k0 + 32, 0, 1);
    // stage B tile transposed: read [k][n], write Bs[n][k]
#pragma unroll
    for (int it = 0; it < 2; ++it) {
      int c = tid + it * 256;
      int kk = c >> 4, nn = (c & 15) * 8;
      int gn = n0 + nn;
      v8bf v = (gn < N) ? *(const v8bf*)(B + (size_t)(k0 + kk) * N + gn)
                        : vzero8bf();
#pragma unroll
      for (int e = 0; e < 8; ++e) Bs[(nn + e) * 32 + kk] = v[e];
    }
    wait_asynccnt0();
    __syncthreads();

    v16bf af[2], bfr[4];
#pragma unroll
    for (int mf = 0; mf < 2; ++mf) {
      int row = wm * 32 + mf * 16 + ln;
      int klo = hl * 8;                              // A layout k-interleave
      af[mf] = comb(*(const v8bf*)&As[row * 32 + klo],
                    *(const v8bf*)&As[row * 32 + klo + 16]);
    }
#pragma unroll
    for (int nf = 0; nf < 4; ++nf) {
      int col = wn * 64 + nf * 16 + ln;
      int ko = hl * 8;                               // B layout k-interleave
      bfr[nf] = comb(*(const v8bf*)&Bs[col * 32 + ko],
                     *(const v8bf*)&Bs[col * 32 + ko + 16]);
    }
#pragma unroll
    for (int mf = 0; mf < 2; ++mf)
#pragma unroll
      for (int nf = 0; nf < 4; ++nf)
        acc[mf][nf] = wmma_bf16(af[mf], bfr[nf], acc[mf][nf]);
    __syncthreads();
  }

#pragma unroll
  for (int mf = 0; mf < 2; ++mf)
#pragma unroll
    for (int nf = 0; nf < 4; ++nf) {
      int col = n0 + wn * 64 + nf * 16 + ln;
      if (col < N) {
#pragma unroll
        for (int r = 0; r < 8; ++r) {
          int row = m0 + wm * 32 + mf * 16 + r + hl * 8;   // C layout
          store_c(C, (size_t)row * N + col, acc[mf][nf][r]);
        }
      }
    }
}

// ---------------------------------------------------------------------------
// RMSNorm over rows of bf16, fp32 weight, bf16 out.
// ---------------------------------------------------------------------------
__global__ __launch_bounds__(256) void rmsnorm_kernel(
    const bf16* __restrict__ in, const float* __restrict__ w,
    bf16* __restrict__ out, int D) {
  const int row = blockIdx.x;
  const bf16* p = in + (size_t)row * D;
  __shared__ float red[8];
  float s = 0.f;
  for (int j = threadIdx.x; j < D; j += blockDim.x) {
    float v = from_bf16(p[j]);
    s += v * v;
  }
#pragma unroll
  for (int d = 1; d < 32; d <<= 1) s += __shfl_xor(s, d, 32);
  if ((threadIdx.x & 31) == 0) red[threadIdx.x >> 5] = s;
  __syncthreads();
  if (threadIdx.x == 0) {
    float t = 0.f;
    for (int i = 0; i < 8; ++i) t += red[i];
    red[0] = t;
  }
  __syncthreads();
  float inv = rsqrtf(red[0] / (float)D + 1e-6f);
  for (int j = threadIdx.x; j < D; j += blockDim.x)
    out[(size_t)row * D + j] = to_bf16(from_bf16(p[j]) * inv * w[j]);
}

// ---------------------------------------------------------------------------
// RoPE: in/out [rows, H*64]; one block per (row, head), 64 threads.
// ---------------------------------------------------------------------------
__global__ __launch_bounds__(64) void rope_kernel(
    const bf16* __restrict__ in, bf16* __restrict__ out, int T, int H) {
  const int bh = blockIdx.x;
  const int head = bh % H;
  const int row = bh / H;
  const int t = row % T;
  const int j = threadIdx.x;   // 0..63
  const size_t base = ((size_t)row * H + head) * 64;
  float x = from_bf16(in[base + j]);
  float other = from_bf16(in[base + ((j < 32) ? (j + 32) : (j - 32))]);
  float rot = (j < 32) ? -other : other;
  float freq = __powf(10000.0f, -(float)(j & 31) / 32.0f);
  float ang = (float)t * freq;
  out[base + j] = to_bf16(x * __cosf(ang) + rot * __sinf(ang));
}

// ---------------------------------------------------------------------------
// Flash attention (causal, online softmax).
//  grid = (T/64, B*16); 128 threads = 4 waves; each wave owns 16 query rows.
//  qn  [B*T, 16*128]  bf16  (nope part, head-major)
//  qpe [B*T, 16*64]   bf16  (rope part)
//  kv  [B*T, 2048]    bf16  cols [0,1024)=k_nop (8 kv-heads x128),
//                           cols [1024,2048)=v (8 kv-heads x128)
//  kpe [B*T, 64]      bf16  shared rope key
//  yo  [B*T, 16*128]  bf16  attention output
// K tile (64 x 192, nope||rope) is loaded by the Tensor Data Mover: two D#
// descriptors with LDS row padding interleave the 128-elem nope rows and the
// 64-elem rope rows into a single [key][192] LDS tile, issued by wave 0 and
// synchronized via s_wait_tensorcnt + workgroup barrier. V tile is staged
// transposed for the WMMA B-operand.
// ---------------------------------------------------------------------------
__global__ __launch_bounds__(128) void mla_attn_kernel(
    const bf16* __restrict__ qn, const bf16* __restrict__ qpe,
    const bf16* __restrict__ kv, const bf16* __restrict__ kpe,
    bf16* __restrict__ yo, int T) {
  __shared__ bf16 Ks[64 * 192];      // K tile row-major [key][192]   (24 KB)
  __shared__ bf16 Vt[128 * 72];      // V tile transposed [dv][key]   (18 KB)
  __shared__ bf16 Pb[4 * 16 * 64];   // per-wave probability tiles    ( 8 KB)

  const int tid = threadIdx.x;
  const int lane = tid & 31;
  const int wid = tid >> 5;
  const int ln = lane & 15;
  const int hl = lane >> 4;
  const int bh = blockIdx.y;
  const int b = bh >> 4;
  const int h = bh & 15;
  const int kvh = h >> 1;            // GQA group = 2
  const int t0 = blockIdx.x * 64;
  const int qrow = t0 + wid * 16;    // wave's first query row
  const float SC = 0.07216878364870322f;  // (128+64)^-0.5

  // ---- preload Q fragments (16 rows x 192 dims = 6 chunks of 32) ----
  v16bf qf[6];
  {
    const int tq = qrow + ln;
#pragma unroll
    for (int dc = 0; dc < 4; ++dc) {
      size_t base = ((size_t)(b * T + tq)) * 2048 + h * 128 + dc * 32 + hl * 8;
      qf[dc] = comb(*(const v8bf*)(qn + base), *(const v8bf*)(qn + base + 16));
    }
#pragma unroll
    for (int dc = 0; dc < 2; ++dc) {
      size_t base = ((size_t)(b * T + tq)) * 1024 + h * 64 + dc * 32 + hl * 8;
      qf[4 + dc] = comb(*(const v8bf*)(qpe + base), *(const v8bf*)(qpe + base + 16));
    }
  }

  v8f Y[8];
#pragma unroll
  for (int vf = 0; vf < 8; ++vf) Y[vf] = vzero8f();
  float mrow[8], lrow[8];
#pragma unroll
  for (int r = 0; r < 8; ++r) { mrow[r] = -3.0e38f; lrow[r] = 0.0f; }

  const int nt = (int)blockIdx.x + 1;       // causal: key tiles 0..blockIdx.x
  for (int kt = 0; kt < nt; ++kt) {
    // ---- TDM-stage K tile (64 keys x 192 = nope||rope) into LDS ----
    if (tid < 32) {
      // nope piece: 64 rows x 128 elems, row stride 2048 elems; pad LDS rows
      // by 64 elems (32 DW) after every 128 elems (code 5 = 256B interval).
      tdm_load_2d(&Ks[0],
                  kv + ((size_t)(b * T + kt * 64)) * 2048 + kvh * 128,
                  128u, 64u, 2048u, /*interval*/5u, /*amount*/31u);
      // rope piece: 64 rows x 64 elems, row stride 64 elems; lands at column
      // 128 of each LDS row; pad by 128 elems (64 DW) after every 64 elems.
      tdm_load_2d(&Ks[128],
                  kpe + ((size_t)(b * T + kt * 64)) * 64,
                  64u, 64u, 64u, /*interval*/4u, /*amount*/63u);
      __builtin_amdgcn_s_wait_tensorcnt(0);
    }
    // prefetch next K tile while this one is consumed
    if (kt + 1 < nt)
      __builtin_prefetch(
          kv + ((size_t)(b * T + (kt + 1) * 64 + (tid >> 1))) * 2048 + kvh * 128,
          0, 1);
    // ---- stage V tile transposed into LDS (all 128 threads) ----
#pragma unroll
    for (int it = 0; it < 8; ++it) {
      int c = tid + it * 128;               // 1024 chunks of 8 dv
      int kk = c >> 4;
      int dvb = (c & 15) * 8;
      size_t gv = ((size_t)(b * T + kt * 64 + kk)) * 2048 + 1024 + kvh * 128 + dvb;
      v8bf v = *(const v8bf*)(kv + gv);
#pragma unroll
      for (int e = 0; e < 8; ++e) Vt[(dvb + e) * 72 + kk] = v[e];
    }
    __syncthreads();

    // ---- S = Q K^T  (K B-fragments from shared LDS tile) ----
    v8f S[4];
#pragma unroll
    for (int nf = 0; nf < 4; ++nf) {
      v8f s = vzero8f();
      const bf16* Kr = &Ks[(nf * 16 + ln) * 192];
#pragma unroll
      for (int dc = 0; dc < 6; ++dc) {
        v16bf kf = comb(*(const v8bf*)&Kr[dc * 32 + hl * 8],
                        *(const v8bf*)&Kr[dc * 32 + hl * 8 + 16]);
        s = wmma_bf16(qf[dc], kf, s);
      }
      S[nf] = s;
    }

    // ---- online softmax (per C-layout row; reduce across 16-lane halves) ----
#pragma unroll
    for (int r = 0; r < 8; ++r) {
      const int qg = qrow + r + hl * 8;
      float mx = -3.0e38f;
#pragma unroll
      for (int nf = 0; nf < 4; ++nf) {
        int kg = kt * 64 + nf * 16 + ln;
        float v = S[nf][r] * SC;
        v = (kg <= qg) ? v : -3.0e38f;
        S[nf][r] = v;
        mx = fmaxf(mx, v);
      }
#pragma unroll
      for (int d = 1; d < 16; d <<= 1) mx = fmaxf(mx, __shfl_xor(mx, d, 32));
      float mn = fmaxf(mrow[r], mx);
      float scl = __expf(mrow[r] - mn);
      mrow[r] = mn;
      float rs = 0.f;
#pragma unroll
      for (int nf = 0; nf < 4; ++nf) {
        float p = __expf(S[nf][r] - mn);
        S[nf][r] = p;
        rs += p;
      }
#pragma unroll
      for (int d = 1; d < 16; d <<= 1) rs += __shfl_xor(rs, d, 32);
      lrow[r] = lrow[r] * scl + rs;
#pragma unroll
      for (int vf = 0; vf < 8; ++vf) Y[vf][r] *= scl;
    }

    // ---- P (C layout) -> LDS -> A layout ----
    bf16* Pw = &Pb[wid * 16 * 64];
#pragma unroll
    for (int nf = 0; nf < 4; ++nf)
#pragma unroll
      for (int r = 0; r < 8; ++r) {
        int m = r + hl * 8;
        Pw[m * 64 + nf * 16 + ln] = to_bf16(S[nf][r]);
      }
    asm volatile("s_wait_dscnt 0" ::: "memory");   // wave-local LDS RAW

    // ---- Y += P * V ----
#pragma unroll
    for (int kc = 0; kc < 2; ++kc) {
      int klo = kc * 32 + hl * 8;                  // A-layout interleave
      v16bf pf = comb(*(const v8bf*)&Pw[ln * 64 + klo],
                      *(const v8bf*)&Pw[ln * 64 + klo + 16]);
#pragma unroll
      for (int vf = 0; vf < 8; ++vf) {
        int dv = vf * 16 + ln;
        int ko = kc * 32 + hl * 8;                 // B layout: k-interleave
        v16bf vfr = comb(*(const v8bf*)&Vt[dv * 72 + ko],
                         *(const v8bf*)&Vt[dv * 72 + ko + 16]);
        Y[vf] = wmma_bf16(pf, vfr, Y[vf]);
      }
    }
    __syncthreads();   // Ks/Vt reused next tile
  }

  // ---- normalize and store ----
#pragma unroll
  for (int vf = 0; vf < 8; ++vf)
#pragma unroll
    for (int r = 0; r < 8; ++r) {
      int trow = qrow + r + hl * 8;
      size_t o = ((size_t)(b * T + trow)) * 2048 + h * 128 + vf * 16 + ln;
      yo[o] = to_bf16(Y[vf][r] / lrow[r]);
    }
}

// ---------------------------------------------------------------------------
// Host launcher
// ---------------------------------------------------------------------------
extern "C" void kernel_launch(void* const* d_in, const int* in_sizes, int n_in,
                              void* d_out, int out_size, void* d_ws, size_t ws_size,
                              hipStream_t stream) {
  (void)n_in; (void)out_size; (void)ws_size;
  const float* x   = (const float*)d_in[0];
  const float* Wqd = (const float*)d_in[1];
  const float* qnw = (const float*)d_in[2];
  const float* Wqu = (const float*)d_in[3];
  const float* Wqp = (const float*)d_in[4];
  const float* Wkd = (const float*)d_in[5];
  const float* kvw = (const float*)d_in[6];
  const float* Wku = (const float*)d_in[7];
  const float* Wkp = (const float*)d_in[8];
  const float* kpw = (const float*)d_in[9];
  const float* Wo  = (const float*)d_in[10];
  float* out = (float*)d_out;

  const int C = 2048;
  const int M = in_sizes[0] / C;     // B*T = 4096
  const int Bsz = 2;
  const int T = M / Bsz;

  char* ws = (char*)d_ws;
  size_t off = 0;
  auto alloc = [&](size_t elems) -> bf16* {
    bf16* p = (bf16*)(ws + off);
    off += (elems * sizeof(bf16) + 255) & ~(size_t)255;
    return p;
  };

  bf16* xb  = alloc((size_t)M * 2048);
  bf16* wqd = alloc((size_t)2048 * 768);
  bf16* wqu = alloc((size_t)768 * 2048);
  bf16* wqp = alloc((size_t)768 * 1024);
  bf16* wkd = alloc((size_t)2048 * 512);
  bf16* wku = alloc((size_t)512 * 2048);
  bf16* wkp = alloc((size_t)2048 * 64);
  bf16* wo  = alloc((size_t)2048 * 2048);
  bf16* tqd = alloc((size_t)M * 768);
  bf16* cq  = alloc((size_t)M * 768);
  bf16* tkv = alloc((size_t)M * 512);
  bf16* ckv = alloc((size_t)M * 512);
  bf16* tkp = alloc((size_t)M * 64);
  bf16* kpn = alloc((size_t)M * 64);
  bf16* kpe = alloc((size_t)M * 64);
  bf16* qnb = alloc((size_t)M * 2048);
  bf16* qpr = alloc((size_t)M * 1024);
  bf16* qpe = alloc((size_t)M * 1024);
  bf16* kvb = alloc((size_t)M * 2048);
  bf16* yo  = alloc((size_t)M * 2048);

  auto cvt = [&](const float* src, bf16* dst, size_t n) {
    int grid = (int)((n + 255) / 256);
    if (grid > 4096) grid = 4096;
    cvt_f32_bf16<<<grid, 256, 0, stream>>>(src, dst, n);
  };
  cvt(x, xb, (size_t)M * 2048);
  cvt(Wqd, wqd, (size_t)2048 * 768);
  cvt(Wqu, wqu, (size_t)768 * 2048);
  cvt(Wqp, wqp, (size_t)768 * 1024);
  cvt(Wkd, wkd, (size_t)2048 * 512);
  cvt(Wku, wku, (size_t)512 * 2048);
  cvt(Wkp, wkp, (size_t)2048 * 64);
  cvt(Wo, wo, (size_t)2048 * 2048);

  auto gemm_b = [&](const bf16* A, const bf16* B, bf16* Co, int m, int n, int k) {
    dim3 g((n + 127) / 128, m / 128);
    gemm_bf16_kernel<bf16><<<g, 256, 0, stream>>>(A, B, Co, m, n, k);
  };
  auto gemm_f = [&](const bf16* A, const bf16* B, float* Co, int m, int n, int k) {
    dim3 g((n + 127) / 128, m / 128);
    gemm_bf16_kernel<float><<<g, 256, 0, stream>>>(A, B, Co, m, n, k);
  };

  // down projections
  gemm_b(xb, wqd, tqd, M, 768, 2048);
  gemm_b(xb, wkd, tkv, M, 512, 2048);
  gemm_b(xb, wkp, tkp, M, 64, 2048);
  // rms norms
  rmsnorm_kernel<<<M, 256, 0, stream>>>(tqd, qnw, cq, 768);
  rmsnorm_kernel<<<M, 256, 0, stream>>>(tkv, kvw, ckv, 512);
  rmsnorm_kernel<<<M, 256, 0, stream>>>(tkp, kpw, kpn, 64);
  // up projections
  gemm_b(cq, wqu, qnb, M, 2048, 768);
  gemm_b(cq, wqp, qpr, M, 1024, 768);
  gemm_b(ckv, wku, kvb, M, 2048, 512);
  // rope
  rope_kernel<<<M * 16, 64, 0, stream>>>(qpr, qpe, T, 16);
  rope_kernel<<<M, 64, 0, stream>>>(kpn, kpe, T, 1);
  // attention
  dim3 ga(T / 64, Bsz * 16);
  mla_attn_kernel<<<ga, 128, 0, stream>>>(qnb, qpe, kvb, kpe, yo, T);
  // output projection -> fp32 out
  gemm_f(yo, wo, out, M, 2048, 2048);
}
